// _Seq2Seq_10677288698280
// MI455X (gfx1250) — matrix operand order repaired and checked by reading, not compiled, not run on hardware
//
#include <hip/hip_runtime.h>
#include <hip/hip_bf16.h>

// ---------------------------------------------------------------------------
// Seq2Seq (LSTM encoder + attention decoder) for MI455X / gfx1250.
// All GEMMs use v_wmma_f32_16x16x32_f16 (f16 inputs, f32 accumulate).
// B=32, S=512, IN=256, H=512, 4H=2048, V+1=32001 (padded to 32064), T=100.
// ---------------------------------------------------------------------------

typedef __attribute__((ext_vector_type(16))) _Float16 v16h;
typedef __attribute__((ext_vector_type(8)))  _Float16 v8h;
typedef __attribute__((ext_vector_type(8)))  float    v8f;

#define DI __device__ __forceinline__

static constexpr int Bz  = 32;
static constexpr int S   = 512;
static constexpr int IN  = 256;
static constexpr int H   = 512;
static constexpr int H4  = 2048;
static constexpr int V1  = 32001;
static constexpr int VP  = 32064;   // 501 * 64, padded vocab
static constexpr int T   = 100;
static constexpr int KE  = IN + H;      // 768  encoder concat K
static constexpr int KD  = 2 * H + H;   // 1536 decoder concat K

DI float sigmoidf_(float x) { return 1.0f / (1.0f + __expf(-x)); }

DI v16h cat16(v8h lo, v8h hi) {
  v16h r;
#pragma unroll
  for (int i = 0; i < 8; ++i) { r[i] = lo[i]; r[i + 8] = hi[i]; }
  return r;
}

// Accumulate one K-region with purely affine addressing (no per-iteration
// source select). kCount must be a multiple of 32.
// aRow0/aRow1: per-lane A pointers (rows rA and rA+16) already offset by the
// lane's koff sub-offset. bRow: per-lane B pointer already offset by kbOff.
DI void wmma_region(const _Float16* __restrict__ aRow0,
                    const _Float16* __restrict__ aRow1,
                    const _Float16* __restrict__ bRow,
                    int kCount, v8f& acc0, v8f& acc1) {
  for (int kt = 0; kt < kCount; kt += 32) {
    v8h a0lo = *(const v8h*)(aRow0 + kt);
    v8h a0hi = *(const v8h*)(aRow0 + kt + 16);
    v8h a1lo = *(const v8h*)(aRow1 + kt);
    v8h a1hi = *(const v8h*)(aRow1 + kt + 16);
    v16h aF0 = cat16(a0lo, a0hi);
    v16h aF1 = cat16(a1lo, a1hi);
    v16h bF  = cat16(*(const v8h*)(bRow + kt), *(const v8h*)(bRow + kt + 16));
    acc0 = __builtin_amdgcn_wmma_f32_16x16x32_f16(false, aF0, false, bF,
                                                  (short)0, acc0, false, false);
    acc1 = __builtin_amdgcn_wmma_f32_16x16x32_f16(false, aF1, false, bF,
                                                  (short)0, acc1, false, false);
  }
}

// ---------------------------------------------------------------------------
// Generic NT WMMA GEMM:  C[32 x N] = A[32 x K] * Bw[N x K]^T (+ bias)
// A is f16, split into two row-major sources at K0 (K0 and Ktot multiples of
// 32, so each source gets its own clean loop). Bw row-major [N, K].
// Block: 128 threads = 4 waves; each wave owns one 16-col tile, both 16-row
// tiles (M=32). Block covers 64 columns; grid.x = N / 64.
// ---------------------------------------------------------------------------
__global__ void k_wmma_gemm_nt(const _Float16* __restrict__ A0, int ldA0, int K0,
                               const _Float16* __restrict__ A1, int ldA1,
                               int Ktot,
                               const _Float16* __restrict__ Bw,
                               float* __restrict__ C, int ldc,
                               const float* __restrict__ bias, int biasN) {
  const int lane  = threadIdx.x & 31;
  const int wave  = threadIdx.x >> 5;
  const int nBase = blockIdx.x * 64 + wave * 16;

  const int rA    = lane & 15;         // row inside 16-row tile
  const int koff  = (lane >> 4) * 8;   // A-frag K sub-offset (ISA 16-bit A layout)
  const int kbOff = (lane >> 4) * 8;   // B-frag K sub-offset

  v8f acc0 = {};  // rows 0..15
  v8f acc1 = {};  // rows 16..31

  const _Float16* bRow = Bw + (size_t)(nBase + (lane & 15)) * Ktot + kbOff;

  // Region 0: K = [0, K0) from A0
  wmma_region(A0 + (size_t)rA * ldA0 + koff,
              A0 + (size_t)(rA + 16) * ldA0 + koff,
              bRow, K0, acc0, acc1);
  // Region 1: K = [K0, Ktot) from A1 (empty when K0 == Ktot)
  if (Ktot > K0) {
    wmma_region(A1 + (size_t)rA * ldA1 + koff,
                A1 + (size_t)(rA + 16) * ldA1 + koff,
                bRow + K0, Ktot - K0, acc0, acc1);
  }

  const int r = lane >> 4;
  const int n = nBase + (lane & 15);
  float bb = 0.0f;
  if (bias) bb = (n < biasN) ? bias[n] : 0.0f;
#pragma unroll
  for (int v = 0; v < 8; ++v) {
    const int m = v + 8 * r;   // C/D layout: VGPR v -> row v (lanes 0-15) / v+8 (16-31)
    C[(size_t)m * ldc + n]        = acc0[v] + bb;
    C[(size_t)(m + 16) * ldc + n] = acc1[v] + bb;
  }
}

// ---------------------------------------------------------------------------
// Setup / conversion kernels
// ---------------------------------------------------------------------------
__global__ void k_cvt_f16(const float* __restrict__ src, _Float16* __restrict__ dst, int n) {
  int i = blockIdx.x * blockDim.x + threadIdx.x;
  if (i < n) dst[i] = (_Float16)src[i];
}

// Wcat_e[2048][768] = [W_ih_e (2048x256) | W_hh_e (2048x512)]  (f16)
__global__ void k_build_we(const float* __restrict__ Wih, const float* __restrict__ Whh,
                           _Float16* __restrict__ W) {
  int i = blockIdx.x * blockDim.x + threadIdx.x;
  if (i >= H4 * KE) return;
  int g = i / KE, k = i % KE;
  float v = (k < IN) ? Wih[g * IN + k] : Whh[g * H + (k - IN)];
  W[i] = (_Float16)v;
}

// Wcat_d[2048][1536] = [W_ih_d (2048x1024) | W_hh_d (2048x512)]  (f16)
__global__ void k_build_wd(const float* __restrict__ Wih, const float* __restrict__ Whh,
                           _Float16* __restrict__ W) {
  int i = blockIdx.x * blockDim.x + threadIdx.x;
  if (i >= H4 * KD) return;
  int g = i / KD, k = i % KD;
  float v = (k < 2 * H) ? Wih[g * (2 * H) + k] : Whh[g * H + (k - 2 * H)];
  W[i] = (_Float16)v;
}

// W_out padded [32064][512]; rows >= 32001 zeroed.
__global__ void k_cvt_wout(const float* __restrict__ W, _Float16* __restrict__ Wp) {
  int i = blockIdx.x * blockDim.x + threadIdx.x;
  if (i >= VP * H) return;
  int n = i / H, k = i % H;
  Wp[i] = (_Float16)((n < V1) ? W[(size_t)n * H + k] : 0.0f);
}

__global__ void k_init_state(float* __restrict__ c, _Float16* __restrict__ h16) {
  int i = blockIdx.x * blockDim.x + threadIdx.x;
  if (i < Bz * H) { c[i] = 0.0f; h16[i] = (_Float16)0.0f; }
}

// dec_in[b][1536]: seg0 (0..511)=inp.h, seg1 (512..1023)=inp.ctx, seg2 (1024..1535)=carry h
__global__ void k_init_dec(_Float16* __restrict__ dec_in, const _Float16* __restrict__ h16) {
  int i = blockIdx.x * blockDim.x + threadIdx.x;
  if (i >= Bz * KD) return;
  int b = i / KD, s = i % KD;
  dec_in[i] = (s < 2 * H) ? (_Float16)0.0f : h16[b * H + (s - 2 * H)];
}

// ---------------------------------------------------------------------------
// LSTM pointwise kernels (gate order i, f, g, o)
// ---------------------------------------------------------------------------
__global__ void k_lstm_enc(const float* __restrict__ gbuf,
                           const float* __restrict__ b_ih, const float* __restrict__ b_hh,
                           float* __restrict__ c, _Float16* __restrict__ h16,
                           float* __restrict__ enc_out, int t) {
  int idx = blockIdx.x * blockDim.x + threadIdx.x;
  if (idx >= Bz * H) return;
  int b = idx >> 9, j = idx & (H - 1);
  const float* g = gbuf + (size_t)b * H4;
  float gi = g[j]            + b_ih[j]            + b_hh[j];
  float gf = g[H + j]        + b_ih[H + j]        + b_hh[H + j];
  float gg = g[2 * H + j]    + b_ih[2 * H + j]    + b_hh[2 * H + j];
  float go = g[3 * H + j]    + b_ih[3 * H + j]    + b_hh[3 * H + j];
  float cN = sigmoidf_(gf) * c[idx] + sigmoidf_(gi) * tanhf(gg);
  float h  = sigmoidf_(go) * tanhf(cN);
  c[idx]   = cN;
  h16[idx] = (_Float16)h;
  enc_out[((size_t)b * S + t) * H + j] = h;
}

__global__ void k_lstm_dec(const float* __restrict__ gbuf,
                           const float* __restrict__ b_ih, const float* __restrict__ b_hh,
                           float* __restrict__ c, float* __restrict__ hbuf,
                           _Float16* __restrict__ dec_in) {
  int idx = blockIdx.x * blockDim.x + threadIdx.x;
  if (idx >= Bz * H) return;
  int b = idx >> 9, j = idx & (H - 1);
  const float* g = gbuf + (size_t)b * H4;
  float gi = g[j]            + b_ih[j]            + b_hh[j];
  float gf = g[H + j]        + b_ih[H + j]        + b_hh[H + j];
  float gg = g[2 * H + j]    + b_ih[2 * H + j]    + b_hh[2 * H + j];
  float go = g[3 * H + j]    + b_ih[3 * H + j]    + b_hh[3 * H + j];
  float cN = sigmoidf_(gf) * c[idx] + sigmoidf_(gi) * tanhf(gg);
  float h  = sigmoidf_(go) * tanhf(cN);
  c[idx]    = cN;
  hbuf[idx] = h;
  _Float16 hh = (_Float16)h;
  dec_in[(size_t)b * KD + j]           = hh;  // next-step inp.h
  dec_in[(size_t)b * KD + 2 * H + j]   = hh;  // carry h (used by logits GEMM + next gates)
}

// ---------------------------------------------------------------------------
// LogSoftmax over the 32001 real vocab entries; writes d_out[b, t, :]
// ---------------------------------------------------------------------------
__global__ void k_logsoftmax(const float* __restrict__ logits, float* __restrict__ out, int t) {
  __shared__ float red[256];
  const int b = blockIdx.x, tid = threadIdx.x;
  const float* row = logits + (size_t)b * VP;

  float mx = -3.402823466e+38f;
  for (int v = tid; v < V1; v += 256) mx = fmaxf(mx, row[v]);
  red[tid] = mx; __syncthreads();
  for (int s = 128; s > 0; s >>= 1) { if (tid < s) red[tid] = fmaxf(red[tid], red[tid + s]); __syncthreads(); }
  mx = red[0]; __syncthreads();

  float sum = 0.0f;
  for (int v = tid; v < V1; v += 256) sum += __expf(row[v] - mx);
  red[tid] = sum; __syncthreads();
  for (int s = 128; s > 0; s >>= 1) { if (tid < s) red[tid] += red[tid + s]; __syncthreads(); }
  const float lse = __logf(red[0]) + mx;

  float* o = out + ((size_t)b * T + t) * V1;
  for (int v = tid; v < V1; v += 256) o[v] = row[v] - lse;
}

// ---------------------------------------------------------------------------
// Attention (per batch b): scores = enc_out[b] . h ; softmax over S ;
// ctx = att @ enc_out[b] -> written as f16 into dec_in seg1.
// ---------------------------------------------------------------------------
__global__ void k_attention(const float* __restrict__ enc_out,
                            const float* __restrict__ hbuf,
                            _Float16* __restrict__ dec_in) {
  __shared__ float sh_h[H];
  __shared__ float sh_s[S];
  __shared__ float red[256];
  const int b = blockIdx.x, tid = threadIdx.x;
  const float* E = enc_out + (size_t)b * S * H;

  for (int j = tid; j < H; j += 256) sh_h[j] = hbuf[b * H + j];
  __syncthreads();

  for (int s = tid; s < S; s += 256) {
    const float* er = E + (size_t)s * H;
    float d = 0.0f;
    for (int k = 0; k < H; ++k) d += er[k] * sh_h[k];
    sh_s[s] = d;
  }
  __syncthreads();

  float mx = fmaxf(sh_s[tid], sh_s[tid + 256]);
  red[tid] = mx; __syncthreads();
  for (int s = 128; s > 0; s >>= 1) { if (tid < s) red[tid] = fmaxf(red[tid], red[tid + s]); __syncthreads(); }
  mx = red[0]; __syncthreads();

  float e0 = __expf(sh_s[tid] - mx);
  float e1 = __expf(sh_s[tid + 256] - mx);
  sh_s[tid] = e0; sh_s[tid + 256] = e1;
  red[tid] = e0 + e1; __syncthreads();
  for (int s = 128; s > 0; s >>= 1) { if (tid < s) red[tid] += red[tid + s]; __syncthreads(); }
  const float inv = 1.0f / red[0]; __syncthreads();

  for (int j = tid; j < H; j += 256) {
    float acc = 0.0f;
    for (int s = 0; s < S; ++s) acc += sh_s[s] * E[(size_t)s * H + j];
    dec_in[(size_t)b * KD + H + j] = (_Float16)(acc * inv);  // seg1 = cols 512..1023
  }
}

// ---------------------------------------------------------------------------
// Host driver
// ---------------------------------------------------------------------------
extern "C" void kernel_launch(void* const* d_in, const int* in_sizes, int n_in,
                              void* d_out, int out_size, void* d_ws, size_t ws_size,
                              hipStream_t stream) {
  const float* x     = (const float*)d_in[0];
  const float* Wihe  = (const float*)d_in[1];
  const float* Whhe  = (const float*)d_in[2];
  const float* bihe  = (const float*)d_in[3];
  const float* bhhe  = (const float*)d_in[4];
  const float* Wihd  = (const float*)d_in[5];
  const float* Whhd  = (const float*)d_in[6];
  const float* bihd  = (const float*)d_in[7];
  const float* bhhd  = (const float*)d_in[8];
  const float* Wout  = (const float*)d_in[9];
  const float* bout  = (const float*)d_in[10];
  float* out = (float*)d_out;

  // ---- carve workspace ----
  char* p = (char*)d_ws;
  auto alloc = [&](size_t bytes) -> char* {
    char* r = p; p += (bytes + 255) & ~(size_t)255; return r;
  };
  _Float16* xh      = (_Float16*)alloc((size_t)Bz * S * IN * 2);   // 8 MB
  _Float16* Wcat_e  = (_Float16*)alloc((size_t)H4 * KE * 2);       // 3 MB
  _Float16* Wcat_d  = (_Float16*)alloc((size_t)H4 * KD * 2);       // 6 MB
  _Float16* Wout_h  = (_Float16*)alloc((size_t)VP * H * 2);        // 32.8 MB
  _Float16* h16     = (_Float16*)alloc((size_t)Bz * H * 2);
  _Float16* dec_in  = (_Float16*)alloc((size_t)Bz * KD * 2);
  float*    cbuf    = (float*)alloc((size_t)Bz * H * 4);
  float*    hbuf    = (float*)alloc((size_t)Bz * H * 4);
  float*    enc_out = (float*)alloc((size_t)Bz * S * H * 4);       // 33.5 MB
  float*    gbuf    = (float*)alloc((size_t)Bz * H4 * 4);
  float*    logits  = (float*)alloc((size_t)Bz * VP * 4);          // 4.1 MB

  // ---- one-time (per call) setup ----
  {
    int n = Bz * S * IN;
    k_cvt_f16<<<(n + 255) / 256, 256, 0, stream>>>(x, xh, n);
  }
  k_build_we<<<(H4 * KE + 255) / 256, 256, 0, stream>>>(Wihe, Whhe, Wcat_e);
  k_build_wd<<<(H4 * KD + 255) / 256, 256, 0, stream>>>(Wihd, Whhd, Wcat_d);
  k_cvt_wout<<<(VP * H + 255) / 256, 256, 0, stream>>>(Wout, Wout_h);
  k_init_state<<<(Bz * H + 255) / 256, 256, 0, stream>>>(cbuf, h16);

  // ---- encoder: 512 sequential steps, 2 kernels each ----
  for (int t = 0; t < S; ++t) {
    // gates[32,2048] = [x_t | h] @ [W_ih_e | W_hh_e]^T   (K = 768)
    k_wmma_gemm_nt<<<H4 / 64, 128, 0, stream>>>(
        xh + (size_t)t * IN, S * IN, IN,   // A0 = x_t (row stride S*IN)
        h16, H,                            // A1 = h
        KE, Wcat_e, gbuf, H4, nullptr, 0);
    k_lstm_enc<<<(Bz * H) / 256, 256, 0, stream>>>(gbuf, bihe, bhhe, cbuf, h16, enc_out, t);
  }

  // ---- decoder init: inp0 = zeros(2H), carry h = hT ----
  k_init_dec<<<(Bz * KD + 255) / 256, 256, 0, stream>>>(dec_in, h16);

  // ---- decoder: 100 sequential steps, 5 kernels each ----
  for (int t = 0; t < T; ++t) {
    // gates[32,2048] = [inp | h] @ [W_ih_d | W_hh_d]^T   (K = 1536)
    k_wmma_gemm_nt<<<H4 / 64, 128, 0, stream>>>(
        dec_in, KD, KD, dec_in, KD, KD, Wcat_d, gbuf, H4, nullptr, 0);
    k_lstm_dec<<<(Bz * H) / 256, 256, 0, stream>>>(gbuf, bihd, bhhd, cbuf, hbuf, dec_in);
    // logits[32, 32064] = h @ W_out^T + b_out   (K = 512)
    k_wmma_gemm_nt<<<VP / 64, 128, 0, stream>>>(
        dec_in + 2 * H, KD, H,             // A0 = h (f16, row stride KD)
        dec_in + 2 * H, KD,                // A1 unused (single source)
        H, Wout_h, logits, VP, bout, V1);
    k_logsoftmax<<<Bz, 256, 0, stream>>>(logits, out, t);
    k_attention<<<Bz, 256, 0, stream>>>(enc_out, hbuf, dec_in);
  }
}
